// Time_LSTM_2_91027536871502
// MI455X (gfx1250) — matrix operand order repaired and checked by reading, not compiled, not run on hardware
//
#include <hip/hip_runtime.h>
#include <hip/hip_bf16.h>
#include <cstdint>
#include <cstddef>

// Problem dims (Time-LSTM): B=16384, IN=512, H=1024, T=64
#define B_  16384
#define IN_ 512
#define H_  1024
#define T_  64

typedef __attribute__((ext_vector_type(16))) __bf16 v16bf;
typedef __attribute__((ext_vector_type(8)))  float  v8f;
typedef __attribute__((ext_vector_type(4)))  unsigned int u32x4;
typedef __attribute__((ext_vector_type(8)))  int i32x8;
typedef __attribute__((ext_vector_type(4)))  int i32x4;

// ---------------------------------------------------------------------------
// helpers
// ---------------------------------------------------------------------------
static __device__ __forceinline__ uint32_t f2bf(float f) {
  union { float f; uint32_t u; } c; c.f = f;
  uint32_t u = c.u;
  uint32_t r = u + 0x7FFFu + ((u >> 16) & 1u);   // round-to-nearest-even
  return r >> 16;
}

static __device__ __forceinline__ float sigf(float x) {
  return 1.0f / (1.0f + __expf(-x));
}

union Frag { uint4 q[2]; v16bf v; };

// Packed fragment = 256 dwords (32 lanes * 8 dwords) = 64 uint4, contiguous.
static __device__ __forceinline__ void load_frag(Frag& f, const uint4* base,
                                                 int frag_id, int lane) {
  const uint4* p = base + (size_t)frag_id * 64 + lane * 2;
  f.q[0] = p[0];
  f.q[1] = p[1];
}

// Fragment read from LDS (emits ds_load_b128 x2).
static __device__ __forceinline__ void lds_frag(Frag& f, const uint4* lb,
                                                int frag_id, int lane) {
  const uint4* p = lb + frag_id * 64 + lane * 2;
  f.q[0] = p[0];
  f.q[1] = p[1];
}

#define WMMA_BF16(ACC, AF, BF)                                                \
  (ACC) = __builtin_amdgcn_wmma_f32_16x16x32_bf16(                            \
      false, (AF), false, (BF), (short)0, (ACC), false, false)

// ---------------------------------------------------------------------------
// Tensor Data Mover: stage `nrows` weight fragments (1KB each, constant
// stride of row_stride_dw dwords) into LDS as one 2-D tile.
// D# packing per CDNA5 ISA (groups 0/1; rest zero).
// ---------------------------------------------------------------------------
static __device__ __forceinline__ void tdm_issue(const uint32_t* gsrc,
                                                 uint32_t lds_byte,
                                                 int nrows, int row_stride_dw) {
  uint64_t ga = (uint64_t)(uintptr_t)gsrc;
  u32x4 g0;
  g0.x = 1u;                                                 // count=1, user D#
  g0.y = lds_byte;                                           // lds_addr
  g0.z = (uint32_t)ga;                                       // global_addr[31:0]
  g0.w = ((uint32_t)(ga >> 32) & 0x01FFFFFFu) | 0x80000000u; // addr[56:32]|type=2
  uint32_t d0 = 256;                 // tile/tensor dim0 (dwords per fragment)
  uint32_t d1 = (uint32_t)nrows;     // rows = fragments
  uint32_t st = (uint32_t)row_stride_dw;
  i32x8 g1;
  g1[0] = (int)(2u << 16);                              // data_size = 4 bytes
  g1[1] = (int)((d0 & 0xFFFFu) << 16);                  // tensor_dim0[15:0]
  g1[2] = (int)((d0 >> 16) | ((d1 & 0xFFFFu) << 16));   // td0[31:16] | td1[15:0]
  g1[3] = (int)((d1 >> 16) | (d0 << 16));               // td1[31:16] | tile_dim0
  g1[4] = (int)(d1 & 0xFFFFu);                          // tile_dim1 (tile_dim2=0)
  g1[5] = (int)st;                                      // tensor_dim0_stride lo
  g1[6] = 0;                                            // stride hi | td1_stride lo
  g1[7] = 0;
  i32x4 z4 = {0, 0, 0, 0};
  i32x8 z8 = {0, 0, 0, 0, 0, 0, 0, 0};
  __builtin_amdgcn_tensor_load_to_lds(g0, g1, z4, z4, z8, 0);
}

#define TDM_STAGE(WBASE, KT, KTIDX, BUF, NG)                                   \
  do {                                                                         \
    if (threadIdx.x < 32)                                                      \
      tdm_issue((WBASE) + ((size_t)nt * (KT) + (KTIDX)) * 256,                 \
                lds_base + (uint32_t)(BUF) * 6144u, (NG), 64 * (KT) * 256);    \
  } while (0)

// `MORE` is a runtime bool; each builtin call carries a constant immediate.
#define TDM_WAIT(MORE)                                                         \
  do {                                                                         \
    if (threadIdx.x < 32) {                                                    \
      if (MORE) __builtin_amdgcn_s_wait_tensorcnt(1);                          \
      else      __builtin_amdgcn_s_wait_tensorcnt(0);                          \
    }                                                                          \
    __syncthreads();                                                           \
  } while (0)

// ---------------------------------------------------------------------------
// Pack fp32 activation matrix [rows x K] row-major into bf16 A-fragments.
// Tile = 16 rows (M) x 32 cols (K).  frag_id = mt*(K/32)+kt, 256 dwords each.
// ISA 16-bit A layout: lane l (group g=l>>4, m=l&15):
//   dword d holds K = kt*32 + (d&4 ? 16:0) + g*8 + (d&3)*2  (two bf16)
// ---------------------------------------------------------------------------
__global__ void pack_a(const float* __restrict__ src, uint32_t* __restrict__ dst,
                       int rows, int K) {
  int idx = blockIdx.x * blockDim.x + threadIdx.x;    // one packed dword
  int total = (rows * K) >> 1;
  if (idx >= total) return;
  int d    = idx & 7;
  int lane = (idx >> 3) & 31;
  int frag = idx >> 8;
  int Kt = K >> 5;
  int kt = frag % Kt, mt = frag / Kt;
  int m = mt * 16 + (lane & 15);
  int g = lane >> 4;
  int k = kt * 32 + ((d & 4) ? 16 : 0) + g * 8 + (d & 3) * 2;
  const float* s = src + (size_t)m * K + k;
  dst[idx] = f2bf(s[0]) | (f2bf(s[1]) << 16);
}

// ---------------------------------------------------------------------------
// Pack fp32 weight matrix [K x N] row-major into bf16 B-fragments.
// Tile = 32 K x 16 N.  frag_id = nt*(K/32)+kt.
// ISA 16-bit B layout (32x16): lane l (g=l>>4, n=l&15):
//   dword d holds K = kt*32 + (d&4 ? 16:0) + g*8 + (d&3)*2, +1.
// ---------------------------------------------------------------------------
__global__ void pack_b(const float* __restrict__ src, uint32_t* __restrict__ dst,
                       int K, int N) {
  int idx = blockIdx.x * blockDim.x + threadIdx.x;
  int total = (K * N) >> 1;
  if (idx >= total) return;
  int d    = idx & 7;
  int lane = (idx >> 3) & 31;
  int frag = idx >> 8;
  int Kt = K >> 5;
  int kt = frag % Kt, nt = frag / Kt;
  int n = nt * 16 + (lane & 15);
  int k = kt * 32 + ((d & 4) ? 16 : 0) + (lane >> 4) * 8 + (d & 3) * 2;
  const float* s = src + (size_t)k * N + n;
  dst[idx] = f2bf(s[0]) | (f2bf(s[N]) << 16);     // (k,n) and (k+1,n)
}

// ---------------------------------------------------------------------------
// Fused gate kernel.  Block = 8 waves sharing one N-tile (nt), spanning 8
// consecutive M-tiles.  Weight B-fragments for the block are staged into LDS
// by the Tensor Data Mover (double-buffered, one DMA in flight), consumed by
// all 8 waves via ds_load_b128.  8 shared accumulators:
//   a_i  = x@Wxi + h@Whi + c@Wci        a_f  = x@Wxf + h@Whf + c@Wcf
//   a_x1 = x@Wxt1                       a_x2 = x@Wxt2
//   a_k  = x@Wxc + h@Whc                a_o  = x@Wxo + h@Who + d@Wto
//   a_t1 = d@Wt1                        a_t2 = d@Wt2
// Epilogue emits c_next (final), c_curl (staged into d_out lo half), o_pre.
// ---------------------------------------------------------------------------
__global__ void __launch_bounds__(256)
lstm_gates(const uint4* __restrict__ Xp,  const uint4* __restrict__ Hp,
           const uint4* __restrict__ Cp,  const uint4* __restrict__ Dp,
           const uint32_t* __restrict__ Wxp, const uint32_t* __restrict__ Whp,
           const uint32_t* __restrict__ Wcp, const uint32_t* __restrict__ Wtp,
           const float* __restrict__ bx,  const float* __restrict__ c_prev,
           float* __restrict__ c_curl_out, float* __restrict__ c_next_out,
           float* __restrict__ o_pre) {
  __shared__ uint32_t ldsB[2 * 1536];            // 2 buffers x 6 fragments x 1KB
  int lane = threadIdx.x & 31;
  int wave = threadIdx.x >> 5;
  int nt = blockIdx.x & 63;                      // H/16 tile, shared by block
  int mt = (blockIdx.x >> 6) * 8 + wave;         // B/16 tile, per wave
  uint32_t lds_base = (uint32_t)(uintptr_t)&ldsB[0];

  v8f a_i = {}, a_f = {}, a_x1 = {}, a_x2 = {}, a_k = {}, a_o = {};
  v8f a_t1 = {}, a_t2 = {};
  Frag A, Bf;
  int buf = 0;

  // ---- x projections: K=512 (Kt=16), six gates of W_x (N=6144) ------------
  TDM_STAGE(Wxp, 16, 0, buf, 6);
  for (int kt = 0; kt < 16; ++kt) {
    load_frag(A, Xp, mt * 16 + kt, lane);
    if (kt + 1 < 16) {
      __builtin_prefetch(Xp + (size_t)(mt * 16 + kt + 1) * 64, 0, 0);
      TDM_STAGE(Wxp, 16, kt + 1, buf ^ 1, 6);
    }
    TDM_WAIT(kt + 1 < 16);
    const uint4* lb = (const uint4*)&ldsB[buf * 1536];
    lds_frag(Bf, lb, 0, lane); WMMA_BF16(a_i,  A.v, Bf.v);
    lds_frag(Bf, lb, 1, lane); WMMA_BF16(a_f,  A.v, Bf.v);
    lds_frag(Bf, lb, 2, lane); WMMA_BF16(a_x1, A.v, Bf.v);
    lds_frag(Bf, lb, 3, lane); WMMA_BF16(a_x2, A.v, Bf.v);
    lds_frag(Bf, lb, 4, lane); WMMA_BF16(a_k,  A.v, Bf.v);
    lds_frag(Bf, lb, 5, lane); WMMA_BF16(a_o,  A.v, Bf.v);
    __syncthreads();
    buf ^= 1;
  }
  // ---- h projections: K=1024 (Kt=32), four gates of W_h (N=4096) ----------
  TDM_STAGE(Whp, 32, 0, buf, 4);
  for (int kt = 0; kt < 32; ++kt) {
    load_frag(A, Hp, mt * 32 + kt, lane);
    if (kt + 1 < 32) TDM_STAGE(Whp, 32, kt + 1, buf ^ 1, 4);
    TDM_WAIT(kt + 1 < 32);
    const uint4* lb = (const uint4*)&ldsB[buf * 1536];
    lds_frag(Bf, lb, 0, lane); WMMA_BF16(a_i, A.v, Bf.v);
    lds_frag(Bf, lb, 1, lane); WMMA_BF16(a_f, A.v, Bf.v);
    lds_frag(Bf, lb, 2, lane); WMMA_BF16(a_k, A.v, Bf.v);
    lds_frag(Bf, lb, 3, lane); WMMA_BF16(a_o, A.v, Bf.v);
    __syncthreads();
    buf ^= 1;
  }
  // ---- c_prev projections: K=1024 (Kt=32), two gates of W_c (N=2048) ------
  TDM_STAGE(Wcp, 32, 0, buf, 2);
  for (int kt = 0; kt < 32; ++kt) {
    load_frag(A, Cp, mt * 32 + kt, lane);
    if (kt + 1 < 32) TDM_STAGE(Wcp, 32, kt + 1, buf ^ 1, 2);
    TDM_WAIT(kt + 1 < 32);
    const uint4* lb = (const uint4*)&ldsB[buf * 1536];
    lds_frag(Bf, lb, 0, lane); WMMA_BF16(a_i, A.v, Bf.v);
    lds_frag(Bf, lb, 1, lane); WMMA_BF16(a_f, A.v, Bf.v);
    __syncthreads();
    buf ^= 1;
  }
  // ---- delt projections: K=64 (Kt=2), three gates of W_t (N=3072) ---------
  TDM_STAGE(Wtp, 2, 0, buf, 3);
  for (int kt = 0; kt < 2; ++kt) {
    load_frag(A, Dp, mt * 2 + kt, lane);
    if (kt + 1 < 2) TDM_STAGE(Wtp, 2, kt + 1, buf ^ 1, 3);
    TDM_WAIT(kt + 1 < 2);
    const uint4* lb = (const uint4*)&ldsB[buf * 1536];
    lds_frag(Bf, lb, 0, lane); WMMA_BF16(a_t1, A.v, Bf.v);
    lds_frag(Bf, lb, 1, lane); WMMA_BF16(a_t2, A.v, Bf.v);
    lds_frag(Bf, lb, 2, lane); WMMA_BF16(a_o,  A.v, Bf.v);
    __syncthreads();
    buf ^= 1;
  }

  // ---- epilogue: C/D tile layout: reg r, lane l -> M = r + 8*(l>>4), N = l&15
  int col     = nt * 16 + (lane & 15);
  int rowbase = mt * 16 + (lane >> 4) * 8;
  float bi = bx[0 * H_ + col], bff = bx[1 * H_ + col], b1 = bx[2 * H_ + col];
  float b2 = bx[3 * H_ + col], bk  = bx[4 * H_ + col], bo = bx[5 * H_ + col];
#pragma unroll
  for (int r = 0; r < 8; ++r) {
    size_t off = (size_t)(rowbase + r) * H_ + col;
    float cp = c_prev[off];
    float i  = sigf(a_i[r] + bi);
    float f  = sigf(a_f[r] + bff);
    float T1 = sigf(a_x1[r] + b1 + sigf(a_t1[r]));
    float T2 = sigf(a_x2[r] + b2 + sigf(a_t2[r]));
    float k  = tanhf(a_k[r] + bk);
    float fc = f * cp;
    float itk = i * k;
    c_curl_out[off] = itk * T1 + fc;
    c_next_out[off] = itk * T2 + fc;
    o_pre[off]      = a_o[r] + bo;
  }
}

// ---------------------------------------------------------------------------
// Output gate GEMM (c_curl @ W_oc) fused with final elementwise.
// Reads c_curl fp32 from d_out lo half and overwrites it with h_next.
// ---------------------------------------------------------------------------
__global__ void __launch_bounds__(256)
out_gate(const uint4* __restrict__ CurlP, const uint4* __restrict__ Wocp,
         const float* __restrict__ o_pre, float* __restrict__ hc) {
  int lane = threadIdx.x & 31;
  int wave = threadIdx.x >> 5;
  int tile = blockIdx.x * (blockDim.x >> 5) + wave;
  int nt = tile & 63;
  int mt = tile >> 6;

  v8f acc = {};
  Frag A, Bf;
  for (int kt = 0; kt < H_ / 32; ++kt) {
    load_frag(A, CurlP, mt * (H_ / 32) + kt, lane);
    if (kt + 1 < H_ / 32)
      __builtin_prefetch(CurlP + (size_t)(mt * (H_ / 32) + kt + 1) * 64, 0, 0);
    load_frag(Bf, Wocp, nt * (H_ / 32) + kt, lane);
    WMMA_BF16(acc, A.v, Bf.v);
  }

  int col     = nt * 16 + (lane & 15);
  int rowbase = mt * 16 + (lane >> 4) * 8;
#pragma unroll
  for (int r = 0; r < 8; ++r) {
    size_t off = (size_t)(rowbase + r) * H_ + col;
    float cc = hc[off];                       // c_curl staged here
    float o  = sigf(o_pre[off] + acc[r]);
    hc[off]  = o * tanhf(cc);                 // h_next overwrites in place
  }
}

// ---------------------------------------------------------------------------
// Host-side launcher
// ---------------------------------------------------------------------------
extern "C" void kernel_launch(void* const* d_in, const int* in_sizes, int n_in,
                              void* d_out, int out_size, void* d_ws, size_t ws_size,
                              hipStream_t stream) {
  (void)in_sizes; (void)n_in; (void)out_size; (void)ws_size;

  const float* x      = (const float*)d_in[0];
  const float* h      = (const float*)d_in[1];
  const float* c_prev = (const float*)d_in[2];
  const float* delt   = (const float*)d_in[3];
  const float* W_x    = (const float*)d_in[4];
  const float* b_x    = (const float*)d_in[5];
  const float* W_h    = (const float*)d_in[6];
  const float* W_c    = (const float*)d_in[7];
  const float* W_t    = (const float*)d_in[8];
  const float* W_oc   = (const float*)d_in[9];

  float* out = (float*)d_out;   // [0,BH) h_next (temp c_curl), [BH,2BH) c_next
  char*  ws  = (char*)d_ws;

  // workspace layout (bytes, all 256B aligned)
  const size_t OFF_XP   = 0;                                  // B*IN*2
  const size_t OFF_HP   = OFF_XP   + (size_t)B_ * IN_ * 2;    // B*H*2
  const size_t OFF_CP   = OFF_HP   + (size_t)B_ * H_ * 2;
  const size_t OFF_DP   = OFF_CP   + (size_t)B_ * H_ * 2;     // B*T*2
  const size_t OFF_WX   = OFF_DP   + (size_t)B_ * T_ * 2;     // IN*6H*2
  const size_t OFF_WH   = OFF_WX   + (size_t)IN_ * 6 * H_ * 2;
  const size_t OFF_WC   = OFF_WH   + (size_t)H_ * 4 * H_ * 2;
  const size_t OFF_WT   = OFF_WC   + (size_t)H_ * 2 * H_ * 2;
  const size_t OFF_WOC  = OFF_WT   + (size_t)T_ * 3 * H_ * 2;
  const size_t OFF_CURL = OFF_WOC  + (size_t)H_ * H_ * 2;     // B*H*2
  const size_t OFF_OPRE = OFF_CURL + (size_t)B_ * H_ * 2;     // B*H*4

  uint32_t* Xp    = (uint32_t*)(ws + OFF_XP);
  uint32_t* Hp    = (uint32_t*)(ws + OFF_HP);
  uint32_t* Cp    = (uint32_t*)(ws + OFF_CP);
  uint32_t* Dp    = (uint32_t*)(ws + OFF_DP);
  uint32_t* Wxp   = (uint32_t*)(ws + OFF_WX);
  uint32_t* Whp   = (uint32_t*)(ws + OFF_WH);
  uint32_t* Wcp   = (uint32_t*)(ws + OFF_WC);
  uint32_t* Wtp   = (uint32_t*)(ws + OFF_WT);
  uint32_t* Wocp  = (uint32_t*)(ws + OFF_WOC);
  uint32_t* CurlP = (uint32_t*)(ws + OFF_CURL);
  float*    Opre  = (float*)   (ws + OFF_OPRE);

  const int TPB = 256;
  auto blocks = [](size_t dwords) { return (unsigned)((dwords + 255) / 256); };

  // 1) pack activations (fp32 -> bf16 A-fragments)
  pack_a<<<blocks((size_t)B_ * IN_ / 2), TPB, 0, stream>>>(x,      Xp, B_, IN_);
  pack_a<<<blocks((size_t)B_ * H_  / 2), TPB, 0, stream>>>(h,      Hp, B_, H_);
  pack_a<<<blocks((size_t)B_ * H_  / 2), TPB, 0, stream>>>(c_prev, Cp, B_, H_);
  pack_a<<<blocks((size_t)B_ * T_  / 2), TPB, 0, stream>>>(delt,   Dp, B_, T_);

  // 2) pack weights (fp32 -> bf16 B-fragments)
  pack_b<<<blocks((size_t)IN_ * 6 * H_ / 2), TPB, 0, stream>>>(W_x,  Wxp,  IN_, 6 * H_);
  pack_b<<<blocks((size_t)H_  * 4 * H_ / 2), TPB, 0, stream>>>(W_h,  Whp,  H_,  4 * H_);
  pack_b<<<blocks((size_t)H_  * 2 * H_ / 2), TPB, 0, stream>>>(W_c,  Wcp,  H_,  2 * H_);
  pack_b<<<blocks((size_t)T_  * 3 * H_ / 2), TPB, 0, stream>>>(W_t,  Wtp,  T_,  3 * H_);
  pack_b<<<blocks((size_t)H_  *     H_ / 2), TPB, 0, stream>>>(W_oc, Wocp, H_,  H_);

  // 3) fused gate WMMA kernel: blocks = (B/16/8) * (H/16) = 8192, 8 waves each
  {
    unsigned nblocks = (B_ / 16 / 8) * (H_ / 16);
    lstm_gates<<<nblocks, TPB, 0, stream>>>(
        (const uint4*)Xp, (const uint4*)Hp, (const uint4*)Cp, (const uint4*)Dp,
        Wxp, Whp, Wcp, Wtp,
        b_x, c_prev,
        out /* c_curl staged */, out + (size_t)B_ * H_ /* c_next */, Opre);
  }

  // 4) repack c_curl (fp32 in d_out lo half) into bf16 A-fragments
  pack_a<<<blocks((size_t)B_ * H_ / 2), TPB, 0, stream>>>(out, CurlP, B_, H_);

  // 5) output-gate GEMM + final elementwise (h_next overwrites c_curl in d_out)
  {
    unsigned ntiles = (B_ / 16) * (H_ / 16);
    out_gate<<<ntiles / 8, TPB, 0, stream>>>(
        (const uint4*)CurlP, (const uint4*)Wocp, Opre, out);
  }
}
